// Model_Mine_78649441124370
// MI455X (gfx1250) — compile-verified, matrix operand order checked by reading, not run on hardware
//
#include <hip/hip_runtime.h>
#include <hip/hip_bf16.h>
#include <math.h>

// ---------------- problem dims (fixed by reference) ----------------
#define NMID 100000   // item table rows
#define D    64       // embedding dim
#define ATT  256      // attention dim (W1 output)
#define NH   4        // num_interest heads
#define CAND 512      // candidate count
#define BATCH 1024
#define SEQ  256
#define NEG_INF (-4294967295.0f)

// LDS row padding (+8 elements) to spread WMMA fragment loads across banks
#define DP   (D + 8)
#define ATTP (ATT + 8)

typedef __attribute__((ext_vector_type(16))) __bf16 v16bf;
typedef __attribute__((ext_vector_type(8)))  float  v8f;

__device__ __forceinline__ __bf16 f2bf(float f) {
  unsigned u = __builtin_bit_cast(unsigned, f);
  unsigned r = u + 0x7FFFu + ((u >> 16) & 1u);      // round-to-nearest-even
  unsigned short h = (unsigned short)(r >> 16);
  return __builtin_bit_cast(__bf16, h);
}
__device__ __forceinline__ float bf2f(__bf16 b) {
  unsigned short h = __builtin_bit_cast(unsigned short, b);
  unsigned u = ((unsigned)h) << 16;
  return __builtin_bit_cast(float, u);
}

// K index inside a v16bf WMMA fragment element e for this lane-half.
// lanes 0-15: K = e (+8 for e>=8 base 16) ; lanes 16-31 offset +8.
__device__ __forceinline__ int frag_k(int e, int half8) {
  return e + half8 + ((e >= 8) ? 8 : 0);
}

// ---------------- stage 0: zero wmean accumulator ----------------
__global__ void mine_zero(float* wmean) { wmean[threadIdx.x] = 0.f; }

// ---------------- stage 1: W1 -> bf16 transposed copy (w1t[n][k], ATT x D) --------
__global__ void mine_w1t(const float* W1, __bf16* w1t) {
  int i = blockIdx.x * 256 + threadIdx.x;       // 256*64 = 16384
  if (i < ATT * D) {
    int a = i >> 6, j = i & 63;                 // a = att col, j = k row
    w1t[i] = f2bf(W1[j * ATT + a]);
  }
}

// ---------------- stage 2: att_w[i,k] = mid_emb[i]·W2[k] ----------------
__global__ __launch_bounds__(256) void mine_attw(const float* mid_emb, const float* W2,
                                                 float* attw) {
  __shared__ float w2s[NH][D];
  int t = threadIdx.x;
  w2s[t >> 6][t & 63] = W2[t];
  __syncthreads();
  int r = blockIdx.x * 256 + t;
  if (r >= NMID) return;
  const float* row = mid_emb + (size_t)r * D;
  float a0 = 0.f, a1 = 0.f, a2 = 0.f, a3 = 0.f;
  for (int j = 0; j < D; ++j) {
    float v = row[j];
    a0 += v * w2s[0][j]; a1 += v * w2s[1][j];
    a2 += v * w2s[2][j]; a3 += v * w2s[3][j];
  }
  attw[r * 4 + 0] = a0; attw[r * 4 + 1] = a1;
  attw[r * 4 + 2] = a2; attw[r * 4 + 3] = a3;
}

// ---------------- stage 3: wmean[k,j] = sum_i attw[i,k]*mid_emb[i,j] ----------------
__global__ __launch_bounds__(256) void mine_wmean(const float* mid_emb, const float* attw,
                                                  float* wmean) {
  int t = threadIdx.x;
  int k = t >> 6, j = t & 63;             // 4*64 = 256 outputs
  int r0 = blockIdx.x * 1024;
  int r1 = r0 + 1024; if (r1 > NMID) r1 = NMID;
  float acc = 0.f;
  for (int r = r0; r < r1; ++r)
    acc += attw[r * 4 + k] * mid_emb[(size_t)r * D + j];
  atomicAdd(&wmean[k * D + j], acc);
}

// ---------------- stage 4: candidates = tanh((W3^T wmean) @ W1), bf16 ----------------
__global__ __launch_bounds__(256) void mine_cand(const float* W3, const float* wmean,
                                                 const float* W1, __bf16* candb) {
  __shared__ float cr[D];
  int i = blockIdx.x, t = threadIdx.x;
  if (t < D) {
    float c = 0.f;
    for (int k = 0; k < NH; ++k) c += W3[k * CAND + i] * wmean[k * D + t];
    cr[t] = c;
  }
  __syncthreads();
  float s = 0.f;
  for (int j = 0; j < D; ++j) s += cr[j] * W1[j * ATT + t];
  candb[(size_t)i * ATT + t] = f2bf(tanhf(s));
}

// ---------------- stage 5: masked mean pooling + mean_hidden (bf16) ----------------
__global__ __launch_bounds__(256) void mine_meanhid(const float* mid_emb, const float* pos,
                                                    const float* W1, const int* his,
                                                    const int* mask, __bf16* mhb) {
  __shared__ int   ids[SEQ];
  __shared__ float msh[SEQ];
  __shared__ float psum[4][D];
  __shared__ float mean[D];
  int b = blockIdx.x, t = threadIdx.x;
  ids[t] = his[(size_t)b * SEQ + t];
  msh[t] = (float)mask[(size_t)b * SEQ + t];
  __syncthreads();
  int g = t >> 6, j = t & 63;
  float acc = 0.f;
  for (int s = g; s < SEQ; s += 4)
    acc += mid_emb[(size_t)ids[s] * D + j] + pos[s * D + j];
  psum[g][j] = acc;
  __syncthreads();
  for (int off = 128; off > 0; off >>= 1) {
    if (t < off) msh[t] += msh[t + off];
    __syncthreads();
  }
  if (t < D)
    mean[t] = (psum[0][t] + psum[1][t] + psum[2][t] + psum[3][t]) / (msh[0] + 1e-9f);
  __syncthreads();
  float s2 = 0.f;
  for (int jj = 0; jj < D; ++jj) s2 += mean[jj] * W1[jj * ATT + t];
  mhb[(size_t)b * ATT + t] = f2bf(tanhf(s2));
}

// ---------------- stage 6: logits = mean_hidden @ candidates^T (WMMA) ----------------
// M=1024, N=512, K=256. Block = 16 rows; 8 waves each own 64 cols.
__global__ __launch_bounds__(256) void mine_logits(const __bf16* mhb, const __bf16* candb,
                                                   float* logits) {
  int m0 = blockIdx.x * 16;
  int w = threadIdx.x >> 5, lane = threadIdx.x & 31;
  int col = lane & 15, half8 = (lane >> 4) * 8;
  v8f zero = {0.f,0.f,0.f,0.f,0.f,0.f,0.f,0.f};
  v8f acc[4] = {zero, zero, zero, zero};
  for (int ks = 0; ks < 8; ++ks) {
    int k0 = ks * 32;
    v16bf av;
#pragma unroll
    for (int e = 0; e < 16; ++e)
      av[e] = mhb[(size_t)(m0 + col) * ATT + k0 + frag_k(e, half8)];
#pragma unroll
    for (int i = 0; i < 4; ++i) {
      int n0 = w * 64 + i * 16;
      v16bf bv;
#pragma unroll
      for (int e = 0; e < 16; ++e)
        bv[e] = candb[(size_t)(n0 + col) * ATT + k0 + frag_k(e, half8)];
      acc[i] = __builtin_amdgcn_wmma_f32_16x16x32_bf16(
          false, av, false, bv, (short)0, acc[i], false, false);
    }
  }
  for (int i = 0; i < 4; ++i) {
    int n0 = w * 64 + i * 16;
#pragma unroll
    for (int r = 0; r < 8; ++r)
      logits[(size_t)(m0 + r + half8) * CAND + n0 + col] = acc[i][r];
  }
}

// ---------------- stage 7: trans gating (argmax -> softmax(trans row) scale) ----------
__global__ __launch_bounds__(256) void mine_trans(const float* trans, float* logits) {
  __shared__ float vals[256];
  __shared__ int   idxs[256];
  __shared__ float tr[CAND];
  __shared__ int   hit_sh;
  int b = blockIdx.x, t = threadIdx.x;
  float* lrow = logits + (size_t)b * CAND;
  float v1 = lrow[t], v2 = lrow[t + 256];
  float bv; int bi;
  if (v1 >= v2) { bv = v1; bi = t; } else { bv = v2; bi = t + 256; }
  vals[t] = bv; idxs[t] = bi;
  __syncthreads();
  for (int off = 128; off > 0; off >>= 1) {
    if (t < off) {
      float ov = vals[t + off]; int oi = idxs[t + off];
      if (ov > vals[t] || (ov == vals[t] && oi < idxs[t])) { vals[t] = ov; idxs[t] = oi; }
    }
    __syncthreads();
  }
  if (t == 0) hit_sh = idxs[0];
  __syncthreads();
  int hit = hit_sh;
  tr[t]       = trans[(size_t)hit * CAND + t];
  tr[t + 256] = trans[(size_t)hit * CAND + 256 + t];
  __syncthreads();
  vals[t] = fmaxf(tr[t], tr[t + 256]);
  __syncthreads();
  for (int off = 128; off > 0; off >>= 1) {
    if (t < off) vals[t] = fmaxf(vals[t], vals[t + off]);
    __syncthreads();
  }
  float mx = vals[0];
  __syncthreads();
  float e1 = expf(tr[t] - mx), e2 = expf(tr[t + 256] - mx);
  vals[t] = e1 + e2;
  __syncthreads();
  for (int off = 128; off > 0; off >>= 1) {
    if (t < off) vals[t] += vals[t + off];
    __syncthreads();
  }
  float Z = vals[0];
  lrow[t]       = v1 * (e1 / Z);
  lrow[t + 256] = v2 * (e2 / Z);
}

// ---------------- stage 8: top-4 indices per row ----------------
__global__ __launch_bounds__(256) void mine_topk(const float* logits, int* topk) {
  __shared__ float vals[256];
  __shared__ int   idxs[256];
  __shared__ int   chosen[NH];
  int b = blockIdx.x, t = threadIdx.x;
  const float* lrow = logits + (size_t)b * CAND;
  for (int it = 0; it < NH; ++it) {
    float bv = -3.4e38f; int bi = 1 << 30;
    for (int c = t; c < CAND; c += 256) {
      bool skip = false;
      for (int p = 0; p < it; ++p) if (chosen[p] == c) skip = true;
      if (skip) continue;
      float v = lrow[c];
      if (v > bv || (v == bv && c < bi)) { bv = v; bi = c; }
    }
    vals[t] = bv; idxs[t] = bi;
    __syncthreads();
    for (int off = 128; off > 0; off >>= 1) {
      if (t < off) {
        float ov = vals[t + off]; int oi = idxs[t + off];
        if (ov > vals[t] || (ov == vals[t] && oi < idxs[t])) { vals[t] = ov; idxs[t] = oi; }
      }
      __syncthreads();
    }
    if (t == 0) chosen[it] = idxs[0];
    __syncthreads();
  }
  if (t < NH) topk[b * NH + t] = chosen[t];
}

// ---------------- stage 9: fused attention per batch row (WMMA x2 in LDS) ----------
__global__ __launch_bounds__(256) void mine_attention(
    const float* mid_emb, const float* pos, const int* his, const int* mids,
    const int* mask, const __bf16* w1t, const __bf16* candb, const int* topk,
    float* out) {
  __shared__ int    ids_sh[SEQ];
  __shared__ __bf16 x_sh[SEQ][DP];         // 36 KB  (A of GEMM1), padded rows
  __shared__ __bf16 item_sh[SEQ][D];       // 32 KB
  __shared__ __bf16 w1_sh[ATT][DP];        // 36 KB  (B of GEMM1, transposed W1)
  __shared__ __bf16 hidden_sh[SEQ][ATTP];  // 132 KB (bf16 tanh(x@W1)), padded rows
  __shared__ __bf16 q_sh[16][ATTP];        // 8.25 KB (padded 4->16 heads)
  __shared__ float  scores_sh[NH][SEQ];    // 4 KB
  __shared__ float  red_sh[256];
  __shared__ float  ue_sh[NH][D];
  __shared__ float  att_sh[NH];
  __shared__ int    sel_sh;

  int b = blockIdx.x, t = threadIdx.x;
  int w = t >> 5, lane = t & 31;
  int col = lane & 15, half8 = (lane >> 4) * 8;

  ids_sh[t] = his[(size_t)b * SEQ + t];
  __syncthreads();

  // gather history embeddings; x = item + pos; stage transposed W1
  for (int f = t; f < SEQ * D; f += 256) {
    int s = f >> 6, j = f & 63;
    float v = mid_emb[(size_t)ids_sh[s] * D + j];
    item_sh[s][j] = f2bf(v);
    x_sh[s][j]    = f2bf(v + pos[s * D + j]);
  }
  for (int f = t; f < ATT * D; f += 256) {
    int a = f >> 6, j = f & 63;
    w1_sh[a][j] = w1t[f];
  }
  // query tile: candidates[topk], padded with zeros to 16 rows
  for (int f = t; f < 16 * ATT; f += 256) {
    int r = f >> 8, a = f & 255;
    __bf16 qv = f2bf(0.f);
    if (r < NH) qv = candb[(size_t)topk[b * NH + r] * ATT + a];
    q_sh[r][a] = qv;
  }
  __syncthreads();

  // GEMM1: hidden = tanh(x @ W1)  (M=256 seq, N=256 att, K=64); 32 tiles/wave
  for (int tt = w * 32; tt < w * 32 + 32; ++tt) {
    int mt = tt >> 4, nt = tt & 15;
    v8f acc = {0.f,0.f,0.f,0.f,0.f,0.f,0.f,0.f};
    for (int ks = 0; ks < 2; ++ks) {
      int k0 = ks * 32;
      v16bf av, bv;
#pragma unroll
      for (int e = 0; e < 16; ++e) {
        int ke = k0 + frag_k(e, half8);
        av[e] = x_sh[mt * 16 + col][ke];
        bv[e] = w1_sh[nt * 16 + col][ke];
      }
      acc = __builtin_amdgcn_wmma_f32_16x16x32_bf16(
          false, av, false, bv, (short)0, acc, false, false);
    }
#pragma unroll
    for (int r = 0; r < 8; ++r)
      hidden_sh[mt * 16 + r + half8][nt * 16 + col] = f2bf(tanhf(acc[r]));
  }
  __syncthreads();

  // GEMM2: scores = Q(16x256) @ hidden^T (K=256 att, N=256 seq); 2 n-tiles/wave
  for (int i = 0; i < 2; ++i) {
    int n0 = (w * 2 + i) * 16;
    v8f acc = {0.f,0.f,0.f,0.f,0.f,0.f,0.f,0.f};
    for (int ks = 0; ks < 8; ++ks) {
      int k0 = ks * 32;
      v16bf av, bv;
#pragma unroll
      for (int e = 0; e < 16; ++e) {
        int ke = k0 + frag_k(e, half8);
        av[e] = q_sh[col][ke];
        bv[e] = hidden_sh[n0 + col][ke];
      }
      acc = __builtin_amdgcn_wmma_f32_16x16x32_bf16(
          false, av, false, bv, (short)0, acc, false, false);
    }
#pragma unroll
    for (int r = 0; r < 8; ++r) {
      int m = r + half8;
      if (m < NH) scores_sh[m][n0 + col] = acc[r];
    }
  }
  __syncthreads();

  // mask + softmax over seq for each head
  float mval = (float)mask[(size_t)b * SEQ + t];
  for (int k = 0; k < NH; ++k) {
    float v = scores_sh[k][t];
    if (mval == 0.f) v = NEG_INF;
    red_sh[t] = v;
    __syncthreads();
    for (int off = 128; off > 0; off >>= 1) {
      if (t < off) red_sh[t] = fmaxf(red_sh[t], red_sh[t + off]);
      __syncthreads();
    }
    float mx = red_sh[0];
    __syncthreads();
    float e = expf(v - mx);
    red_sh[t] = e;
    __syncthreads();
    for (int off = 128; off > 0; off >>= 1) {
      if (t < off) red_sh[t] += red_sh[t + off];
      __syncthreads();
    }
    float Z = red_sh[0];
    __syncthreads();
    scores_sh[k][t] = e / Z;
    __syncthreads();
  }

  // user_eb[k][j] = sum_s scores[k][s] * item[s][j]
  {
    int k = t >> 6, j = t & 63;
    float acc = 0.f;
    for (int s = 0; s < SEQ; ++s) acc += scores_sh[k][s] * bf2f(item_sh[s][j]);
    ue_sh[k][j] = acc;
  }
  __syncthreads();

  // readout: pick head with max affinity to target item
  if (t < NH) {
    const float* ie = mid_emb + (size_t)mids[b] * D;
    float a = 0.f;
    for (int j = 0; j < D; ++j) a += ue_sh[t][j] * ie[j];
    att_sh[t] = a;
  }
  __syncthreads();
  if (t == 0) {
    int best = 0; float bv = att_sh[0];
    for (int k = 1; k < NH; ++k) if (att_sh[k] > bv) { bv = att_sh[k]; best = k; }
    sel_sh = best;
  }
  __syncthreads();
  if (t < D) out[(size_t)b * D + t] = ue_sh[sel_sh][t];
}

// ---------------- host launch ----------------
extern "C" void kernel_launch(void* const* d_in, const int* in_sizes, int n_in,
                              void* d_out, int out_size, void* d_ws, size_t ws_size,
                              hipStream_t stream) {
  (void)in_sizes; (void)n_in; (void)out_size; (void)ws_size;
  const float* mid_emb = (const float*)d_in[0];
  const float* W1      = (const float*)d_in[1];
  const float* W2      = (const float*)d_in[2];
  const float* W3      = (const float*)d_in[3];
  const float* trans   = (const float*)d_in[4];
  const float* pos     = (const float*)d_in[5];
  const int*   his     = (const int*)d_in[6];
  const int*   mids    = (const int*)d_in[7];
  const int*   mask    = (const int*)d_in[8];
  float* out = (float*)d_out;

  char* ws = (char*)d_ws;
  size_t off = 0;
  auto alloc = [&](size_t bytes) -> void* {
    void* p = ws + off;
    off = (off + bytes + 255) & ~(size_t)255;
    return p;
  };
  float*  wmean  = (float*)alloc(NH * D * sizeof(float));
  float*  attw   = (float*)alloc((size_t)NMID * NH * sizeof(float));
  __bf16* candb  = (__bf16*)alloc((size_t)CAND * ATT * sizeof(__bf16));
  __bf16* w1t    = (__bf16*)alloc((size_t)ATT * D * sizeof(__bf16));
  __bf16* mhb    = (__bf16*)alloc((size_t)BATCH * ATT * sizeof(__bf16));
  float*  logits = (float*)alloc((size_t)BATCH * CAND * sizeof(float));
  int*    topk   = (int*)alloc((size_t)BATCH * NH * sizeof(int));

  mine_zero<<<1, 256, 0, stream>>>(wmean);
  mine_w1t<<<(ATT * D + 255) / 256, 256, 0, stream>>>(W1, w1t);
  mine_attw<<<(NMID + 255) / 256, 256, 0, stream>>>(mid_emb, W2, attw);
  mine_wmean<<<(NMID + 1023) / 1024, 256, 0, stream>>>(mid_emb, attw, wmean);
  mine_cand<<<CAND, 256, 0, stream>>>(W3, wmean, W1, candb);
  mine_meanhid<<<BATCH, 256, 0, stream>>>(mid_emb, pos, W1, his, mask, mhb);
  mine_logits<<<BATCH / 16, 256, 0, stream>>>(mhb, candb, logits);
  mine_trans<<<BATCH, 256, 0, stream>>>(trans, logits);
  mine_topk<<<BATCH, 256, 0, stream>>>(logits, topk);
  mine_attention<<<BATCH, 256, 0, stream>>>(mid_emb, pos, his, mids, mask,
                                            w1t, candb, topk, out);
}
